// MultiHeadedAttention_86577950752968
// MI455X (gfx1250) — compile-verified, matrix operand order checked by reading, not run on hardware
//
#include <hip/hip_runtime.h>
#include <hip/hip_bf16.h>

// ---------------------------------------------------------------------------
// MHA for B=2, S=2048, D=1024, H=16, DK=64 on gfx1250 (CDNA5, wave32, WMMA)
// GEMMs + flash attention via v_wmma_f32_16x16x32_bf16; bf16 tile staging via
// the Tensor Data Mover (tensor_load_to_lds) with hardware row padding.
// ---------------------------------------------------------------------------

typedef __attribute__((ext_vector_type(16))) __bf16 v16bf;
typedef __attribute__((ext_vector_type(8)))  __bf16 v8bf;
typedef __attribute__((ext_vector_type(8)))  float  v8f;
typedef unsigned short u16;
typedef unsigned int u32x4 __attribute__((ext_vector_type(4)));
typedef int i32x8 __attribute__((ext_vector_type(8)));
typedef int i32x4 __attribute__((ext_vector_type(4)));

#define B_  2
#define S_  2048
#define H_  16
#define D_  1024
#define DK_ 64
#define MTOT_ (B_ * S_)   // 4096

static __device__ __forceinline__ u16 f2bf(float f) {
    return __builtin_bit_cast(u16, (__bf16)f);
}

static __device__ __forceinline__ v8f wmma_bf16(v16bf a, v16bf b, v8f c) {
    // v_wmma_f32_16x16x32_bf16  D = A(16x32) x B(32x16) + C(16x16 f32)
    return __builtin_amdgcn_wmma_f32_16x16x32_bf16(
        /*neg_a=*/false, a, /*neg_b=*/false, b,
        /*c_mod=*/(short)0, c, /*reuse_a=*/false, /*reuse_b=*/false);
}

// TDM 2-D tile load: global (row-major, row_stride elems of bf16) -> LDS with
// pad_amount dwords appended every pad_interval dwords (i.e. per row).
// D# layout per CDNA5 ISA.
// Must be issued by exactly one wave (TDM ignores EXEC, issues per-wave).
static __device__ __forceinline__ void tdm_load_2d(
    unsigned lds_addr, const void* gaddr,
    unsigned tile_w, unsigned tile_h, unsigned row_stride, unsigned tensor_h,
    unsigned pad_interval_code, unsigned pad_amount_code) {
    unsigned long long ga = (unsigned long long)gaddr;
    u32x4 g0;
    g0[0] = 1u;                                         // count=1, user mode
    g0[1] = lds_addr;                                   // lds_addr [63:32]
    g0[2] = (unsigned)(ga & 0xFFFFFFFFu);               // global_addr low
    g0[3] = (unsigned)((ga >> 32) & 0x01FFFFFFu)        // global_addr[56:32]
          | (2u << 30);                                 // type=2 ("image")
    i32x8 g1;
    g1[0] = (int)((1u << 16)                            // data_size=1 -> 2B
                | (1u << 20)                            // pad_enable
                | (pad_interval_code << 22)
                | (pad_amount_code << 25));
    g1[1] = (int)((row_stride & 0xFFFFu) << 16);        // tensor_dim0[15:0]
    g1[2] = (int)(((row_stride >> 16) & 0xFFFFu)        // tensor_dim0[31:16]
                | ((tensor_h & 0xFFFFu) << 16));        // tensor_dim1[15:0]
    g1[3] = (int)(((tensor_h >> 16) & 0xFFFFu)          // tensor_dim1[31:16]
                | ((tile_w & 0xFFFFu) << 16));          // tile_dim0
    g1[4] = (int)(tile_h & 0xFFFFu);                    // tile_dim1 (tile_dim2=0)
    g1[5] = (int)row_stride;                            // tensor_dim0_stride lo
    g1[6] = 0;                                          // stride hi / dim1_stride lo
    g1[7] = 0;
    i32x4 gz4 = {0, 0, 0, 0};                           // 2-D: groups 2/3 unused
    i32x8 gz8 = {0, 0, 0, 0, 0, 0, 0, 0};
    // amdgpu-toolchain (clang-23) 6-arg form.
    __builtin_amdgcn_tensor_load_to_lds(g0, g1, gz4, gz4, gz8, /*cpol=*/0);
}

static __device__ __forceinline__ unsigned lds_off(const void* p) {
    return (unsigned)(unsigned long long)p;             // LDS offset = addr[31:0]
}

// Load a 16x32 bf16 A-style fragment.
// Layout (ISA 7.12.2): lanes 0-15 hold M=0..15 with K={0..7,16..23},
// lanes 16-31 hold M=0..15 with K={8..15,24..31}. base is row-major with
// `pitch` bf16 elements per row; `row` = tile row base, `koff` = K base.
static __device__ __forceinline__ v16bf ld_frag(const u16* base, int pitch,
                                                int row, int koff, int lane) {
    const u16* p = base + (size_t)(row + (lane & 15)) * pitch + koff
                        + ((lane < 16) ? 0 : 8);
    v8bf lo = *(const v8bf*)p;         // K sel .. sel+7
    v8bf hi = *(const v8bf*)(p + 16);  // K sel+16 .. sel+23
    return __builtin_shufflevector(lo, hi, 0, 1, 2, 3, 4, 5, 6, 7,
                                            8, 9, 10, 11, 12, 13, 14, 15);
}

// ---------------------------------------------------------------------------
// GEMM: C[M,N] = A[M,K] @ W[N,K]^T + bias[N]
// Block tile 128x64, BK=32 (one WMMA k-step), 256 threads = 8 waves (4x2),
// each wave owns a 32x32 accumulator = 2x2 WMMA tiles.
// bf16 A tiles are staged by the TDM; fp32 A/W tiles convert through VALU.
// ---------------------------------------------------------------------------
template <bool A_BF16, bool OUT_BF16>
__global__ __launch_bounds__(256) void gemm_wmma_kernel(
    const void* __restrict__ Araw, const float* __restrict__ W,
    const float* __restrict__ bias, void* __restrict__ Cout,
    int M, int N, int K) {
    constexpr int BM = 128, BN = 64, BK = 32, PITCH = 40;  // 80B rows, 16B-aligned
    __shared__ u16 sA[BM * PITCH];
    __shared__ u16 sB[BN * PITCH];

    const int tid  = threadIdx.x;
    const int lane = tid & 31;
    const int wave = tid >> 5;
    const int wm   = wave >> 1;  // 0..3
    const int wn   = wave & 1;   // 0..1
    const int mbase = blockIdx.x * BM;
    const int nbase = blockIdx.y * BN;

    v8f acc[2][2] = {};

    for (int k0 = 0; k0 < K; k0 += BK) {
        // ---- stage A tile: 128 rows x 32 cols
        if constexpr (A_BF16) {
            // One TDM descriptor moves the whole 128x32 bf16 tile, padding
            // each 16-dword (32-elem) row with 4 dwords -> LDS pitch 40.
            if (wave == 0) {
                tdm_load_2d(lds_off(&sA[0]),
                            (const u16*)Araw + (size_t)(mbase)*K + k0,
                            /*tile_w=*/BK, /*tile_h=*/BM,
                            /*row_stride=*/K, /*tensor_h=*/M - mbase,
                            /*pad_interval(16dw)=*/3, /*pad_amount(4dw)=*/3);
                __builtin_amdgcn_s_wait_tensorcnt(0);
            }
        } else {
            const int r = tid >> 1;
            const int c = (tid & 1) * 16;
            const float* ap = (const float*)Araw + (size_t)(mbase + r) * K + k0 + c;
            float fv[16];
            ((float4*)fv)[0] = ((const float4*)ap)[0];
            ((float4*)fv)[1] = ((const float4*)ap)[1];
            ((float4*)fv)[2] = ((const float4*)ap)[2];
            ((float4*)fv)[3] = ((const float4*)ap)[3];
            v8bf o0, o1;
#pragma unroll
            for (int i = 0; i < 8; ++i) { o0[i] = (__bf16)fv[i]; o1[i] = (__bf16)fv[8 + i]; }
            *(v8bf*)&sA[r * PITCH + c]     = o0;
            *(v8bf*)&sA[r * PITCH + c + 8] = o1;
            if (k0 + BK < K) __builtin_prefetch(ap + BK, 0, 3);
        }
        // ---- stage W tile: 64 rows (output cols) x 32 cols (4 threads / row)
        {
            const int r = tid >> 2;
            const int c = (tid & 3) * 8;
            const float* wp = W + (size_t)(nbase + r) * K + k0 + c;
            float fv[8];
            ((float4*)fv)[0] = ((const float4*)wp)[0];
            ((float4*)fv)[1] = ((const float4*)wp)[1];
            v8bf o;
#pragma unroll
            for (int i = 0; i < 8; ++i) o[i] = (__bf16)fv[i];
            *(v8bf*)&sB[r * PITCH + c] = o;
            if (k0 + BK < K) __builtin_prefetch(wp + BK, 0, 3);
        }
        __syncthreads();

        v16bf af[2], bf[2];
#pragma unroll
        for (int i = 0; i < 2; ++i) af[i] = ld_frag(sA, PITCH, wm * 32 + i * 16, 0, lane);
#pragma unroll
        for (int j = 0; j < 2; ++j) bf[j] = ld_frag(sB, PITCH, wn * 32 + j * 16, 0, lane);
#pragma unroll
        for (int i = 0; i < 2; ++i)
#pragma unroll
            for (int j = 0; j < 2; ++j) acc[i][j] = wmma_bf16(af[i], bf[j], acc[i][j]);
        __syncthreads();
    }

    // ---- epilogue: bias add + store.
    // C layout: VGPR r of lane L -> row = r + (L<16 ? 0 : 8), col = L%16.
#pragma unroll
    for (int i = 0; i < 2; ++i) {
#pragma unroll
        for (int j = 0; j < 2; ++j) {
            const int col  = nbase + wn * 32 + j * 16 + (lane & 15);
            const int rowb = mbase + wm * 32 + i * 16 + ((lane < 16) ? 0 : 8);
            const float bv = bias[col];
#pragma unroll
            for (int r = 0; r < 8; ++r) {
                const float v = acc[i][j][r] + bv;
                const size_t idx = (size_t)(rowb + r) * N + col;
                if constexpr (OUT_BF16) ((u16*)Cout)[idx]   = f2bf(v);
                else                    ((float*)Cout)[idx] = v;
            }
        }
    }
}

// ---------------------------------------------------------------------------
// Flash attention over bf16 Q/K/V (stored [B,S,D], head h = cols h*64..h*64+63).
// Grid: x = S/128 query blocks, y = B*H. 256 threads = 8 waves, each wave owns
// 16 query rows. K chunks staged by the TDM; V chunks transposed into LDS;
// online softmax; per-wave LDS bounce converts the f32 C-layout score tile
// into the bf16 A-layout operand for P@V.
// ---------------------------------------------------------------------------
__global__ __launch_bounds__(256) void flash_attn_kernel(
    const u16* __restrict__ q, const u16* __restrict__ k,
    const u16* __restrict__ v, u16* __restrict__ z) {
    constexpr int QB = 128, KC = 64, PITCH = 72;  // 144B rows, 16B-aligned
    __shared__ u16 sK[KC * PITCH];                // K chunk, row-major [key][d]
    __shared__ u16 sV[DK_ * PITCH];               // V chunk transposed [d][key]
    __shared__ u16 sP[8 * 16 * PITCH];            // per-wave P scratch

    const int tid  = threadIdx.x;
    const int lane = tid & 31;
    const int wave = tid >> 5;
    const int bh   = blockIdx.y;
    const int b    = bh / H_;
    const int h    = bh % H_;
    const int qrow0 = blockIdx.x * QB + wave * 16;

    const u16* qp = q + (size_t)b * S_ * D_ + (size_t)h * DK_;
    const u16* kp = k + (size_t)b * S_ * D_ + (size_t)h * DK_;
    const u16* vp = v + (size_t)b * S_ * D_ + (size_t)h * DK_;

    // Q fragments for this wave's 16 rows (K-dim = 64 -> two 16x32 frags),
    // loaded straight from global in A-fragment layout, kept in VGPRs.
    v16bf qf[2];
#pragma unroll
    for (int ks = 0; ks < 2; ++ks) {
        const u16* p = qp + (size_t)(qrow0 + (lane & 15)) * D_ + ks * 32
                          + ((lane < 16) ? 0 : 8);
        v8bf lo = *(const v8bf*)p;
        v8bf hi = *(const v8bf*)(p + 16);
        qf[ks] = __builtin_shufflevector(lo, hi, 0, 1, 2, 3, 4, 5, 6, 7,
                                                 8, 9, 10, 11, 12, 13, 14, 15);
    }

    float m_r[8], l_r[8];
    v8f zacc[4] = {};
#pragma unroll
    for (int r = 0; r < 8; ++r) { m_r[r] = -1e30f; l_r[r] = 0.0f; }

    u16* pS = &sP[wave * 16 * PITCH];

    for (int kc0 = 0; kc0 < S_; kc0 += KC) {
        // ---- stage V chunk transposed into LDS (all threads)
        {
            const int r = tid >> 2;
            const int c = (tid & 3) * 16;
            const u16* vs = vp + (size_t)(kc0 + r) * D_ + c;
            u16 tmp[16];
            ((uint4*)tmp)[0] = *(const uint4*)(vs);
            ((uint4*)tmp)[1] = *(const uint4*)(vs + 8);
#pragma unroll
            for (int e = 0; e < 16; ++e) sV[(c + e) * PITCH + r] = tmp[e];
            if (kc0 + KC < S_) __builtin_prefetch(vs + (size_t)KC * D_, 0, 3);
        }
        // ---- stage K chunk via TDM: 64x64 bf16 rows (32 dwords) padded by
        //      4 dwords each -> LDS pitch 72 elements.
        if (wave == 0) {
            tdm_load_2d(lds_off(&sK[0]), kp + (size_t)kc0 * D_,
                        /*tile_w=*/DK_, /*tile_h=*/KC,
                        /*row_stride=*/D_, /*tensor_h=*/S_ - kc0,
                        /*pad_interval(32dw)=*/4, /*pad_amount(4dw)=*/3);
            __builtin_amdgcn_s_wait_tensorcnt(0);
        }
        __syncthreads();

        // ---- scores S = (Q @ K^T) * 1/sqrt(DK): four 16x16 tiles over 64 keys
        v8f sc[4];
#pragma unroll
        for (int j = 0; j < 4; ++j) {
            v8f a = {};
#pragma unroll
            for (int ks = 0; ks < 2; ++ks)
                a = wmma_bf16(qf[ks], ld_frag(sK, PITCH, j * 16, ks * 32, lane), a);
            sc[j] = a;
        }
        constexpr float scl = 0.125f;  // 1/sqrt(64)
#pragma unroll
        for (int j = 0; j < 4; ++j)
#pragma unroll
            for (int r = 0; r < 8; ++r) sc[j][r] *= scl;

        // ---- online softmax row statistics (rows live per-VGPR; reduce the
        //      16-lane half each row occupies via xor shuffles 1/2/4/8)
        float alpha[8];
#pragma unroll
        for (int r = 0; r < 8; ++r) {
            float mx = fmaxf(fmaxf(sc[0][r], sc[1][r]), fmaxf(sc[2][r], sc[3][r]));
            mx = fmaxf(mx, __shfl_xor(mx, 1, 32));
            mx = fmaxf(mx, __shfl_xor(mx, 2, 32));
            mx = fmaxf(mx, __shfl_xor(mx, 4, 32));
            mx = fmaxf(mx, __shfl_xor(mx, 8, 32));
            const float mn = fmaxf(m_r[r], mx);
            alpha[r] = __expf(m_r[r] - mn);
            m_r[r]   = mn;
        }

        // ---- P = exp(S - m); write bf16 P to per-wave scratch, fold row sums
        const int Mrow = ((lane < 16) ? 0 : 8);
#pragma unroll
        for (int r = 0; r < 8; ++r) {
            float rs = 0.0f;
#pragma unroll
            for (int j = 0; j < 4; ++j) {
                const float p = __expf(sc[j][r] - m_r[r]);
                rs += p;
                pS[(r + Mrow) * PITCH + j * 16 + (lane & 15)] = f2bf(p);
            }
            rs += __shfl_xor(rs, 1, 32);
            rs += __shfl_xor(rs, 2, 32);
            rs += __shfl_xor(rs, 4, 32);
            rs += __shfl_xor(rs, 8, 32);
            l_r[r] = l_r[r] * alpha[r] + rs;
#pragma unroll
            for (int dt = 0; dt < 4; ++dt) zacc[dt][r] *= alpha[r];
        }

        // ---- Z += P @ V  (P: 16x64 A-frags from scratch; V^T rows are d)
        v16bf pf[2];
#pragma unroll
        for (int ks = 0; ks < 2; ++ks) pf[ks] = ld_frag(pS, PITCH, 0, ks * 32, lane);
#pragma unroll
        for (int dt = 0; dt < 4; ++dt)
#pragma unroll
            for (int ks = 0; ks < 2; ++ks)
                zacc[dt] = wmma_bf16(pf[ks], ld_frag(sV, PITCH, dt * 16, ks * 32, lane),
                                     zacc[dt]);
        __syncthreads();
    }

    // ---- normalize and store bf16 Z back to [B,S,D]
#pragma unroll
    for (int dt = 0; dt < 4; ++dt) {
        const int col = h * DK_ + dt * 16 + (lane & 15);
#pragma unroll
        for (int r = 0; r < 8; ++r) {
            const int srow = qrow0 + r + ((lane < 16) ? 0 : 8);
            const float vz = zacc[dt][r] * (1.0f / l_r[r]);
            z[((size_t)b * S_ + srow) * D_ + col] = f2bf(vz);
        }
    }
}

// ---------------------------------------------------------------------------
extern "C" void kernel_launch(void* const* d_in, const int* in_sizes, int n_in,
                              void* d_out, int out_size, void* d_ws, size_t ws_size,
                              hipStream_t stream) {
    (void)in_sizes; (void)n_in; (void)out_size; (void)ws_size;
    const float* q_in = (const float*)d_in[0];
    const float* k_in = (const float*)d_in[1];
    const float* v_in = (const float*)d_in[2];
    const float* Wq_w = (const float*)d_in[3];
    const float* Wq_b = (const float*)d_in[4];
    const float* Wk_w = (const float*)d_in[5];
    const float* Wk_b = (const float*)d_in[6];
    const float* Wv_w = (const float*)d_in[7];
    const float* Wv_b = (const float*)d_in[8];
    const float* Wo_w = (const float*)d_in[9];
    const float* Wo_b = (const float*)d_in[10];

    // workspace: 4 bf16 [4096 x 1024] buffers = 32 MB
    u16* qb = (u16*)d_ws;
    u16* kb = qb + (size_t)MTOT_ * D_;
    u16* vb = kb + (size_t)MTOT_ * D_;
    u16* zb = vb + (size_t)MTOT_ * D_;

    dim3 blk(256);
    dim3 gemm_grid(MTOT_ / 128, D_ / 64);  // 32 x 16

    // Q/K/V projections: fp32 A, bf16 out
    gemm_wmma_kernel<false, true><<<gemm_grid, blk, 0, stream>>>(
        q_in, Wq_w, Wq_b, qb, MTOT_, D_, D_);
    gemm_wmma_kernel<false, true><<<gemm_grid, blk, 0, stream>>>(
        k_in, Wk_w, Wk_b, kb, MTOT_, D_, D_);
    gemm_wmma_kernel<false, true><<<gemm_grid, blk, 0, stream>>>(
        v_in, Wv_w, Wv_b, vb, MTOT_, D_, D_);

    // flash attention: 16 query blocks x 32 (b,h)
    dim3 attn_grid(S_ / 128, B_ * H_);
    flash_attn_kernel<<<attn_grid, blk, 0, stream>>>(qb, kb, vb, zb);

    // output projection: bf16 A, fp32 out
    gemm_wmma_kernel<true, false><<<gemm_grid, blk, 0, stream>>>(
        zb, Wo_w, Wo_b, (float*)d_out, MTOT_, D_, D_);
}
